// PentachoronCantorAttention_46523085750338
// MI455X (gfx1250) — matrix operand order repaired and checked by reading, not compiled, not run on hardware
//
#include <hip/hip_runtime.h>
#include <hip/hip_bf16.h>
#include <math.h>

typedef __attribute__((ext_vector_type(16))) _Float16 v16h;
typedef __attribute__((ext_vector_type(8)))  _Float16 v8h;
typedef __attribute__((ext_vector_type(8)))  float    v8f;

#define SEQ_N   2048
#define DIM     512
#define QKV3    1536
#define HEADS   8
#define HD      64
#define NCENT   500
#define CPAD    512
#define KWIN    128

// ---------------------------------------------------------------------------
// f32 -> f16 convert
// ---------------------------------------------------------------------------
__global__ void cvt_f16_kernel(const float* __restrict__ src,
                               _Float16* __restrict__ dst, int n) {
    int i = blockIdx.x * blockDim.x + threadIdx.x;
    if (i < n) dst[i] = (_Float16)src[i];
}

// centroids = mean over 5 vertices; rows [500,512) zero-padded
__global__ void centroid_kernel(const float* __restrict__ penta,
                                _Float16* __restrict__ cent_h) {
    int idx = blockIdx.x * blockDim.x + threadIdx.x;   // over CPAD*DIM
    if (idx >= CPAD * DIM) return;
    int c = idx >> 9;          // /512
    int d = idx & (DIM - 1);
    float v = 0.0f;
    if (c < NCENT) {
        for (int i = 0; i < 5; ++i) v += penta[(size_t)(c * 5 + i) * DIM + d];
        v *= 0.2f;
    }
    cent_h[idx] = (_Float16)v;
}

// per-row L2 normalize x -> f16
__global__ void rownorm_kernel(const float* __restrict__ x,
                               _Float16* __restrict__ xn) {
    __shared__ float red[128];
    int n = blockIdx.x, tid = threadIdx.x;   // 128 threads
    float s = 0.0f;
    for (int d = tid; d < DIM; d += 128) {
        float v = x[(size_t)n * DIM + d];
        s += v * v;
    }
    red[tid] = s;
    __syncthreads();
    for (int st = 64; st > 0; st >>= 1) {
        if (tid < st) red[tid] += red[tid + st];
        __syncthreads();
    }
    float inv = 1.0f / (sqrtf(red[0]) + 1e-12f);
    for (int d = tid; d < DIM; d += 128)
        xn[(size_t)n * DIM + d] = (_Float16)(x[(size_t)n * DIM + d] * inv);
}

// ---------------------------------------------------------------------------
// Generic WMMA GEMM: C[M,N] = A[M,K](f16,row-major) * B[N,K](f16,row-major)^T
// + bias[N].  One wave computes one 16x16 tile. K multiple of 32.
// Lane layouts per CDNA5 ISA 7.12.2 (16-bit A 16x32, 16-bit B 32x16,
// 32-bit C/D 16x16).
// ---------------------------------------------------------------------------
__global__ void gemm_f16_wmma(const _Float16* __restrict__ A,
                              const _Float16* __restrict__ B,
                              const float* __restrict__ bias,
                              float* __restrict__ C,
                              int M, int N, int K) {
    int wave = blockIdx.x * (blockDim.x >> 5) + (threadIdx.x >> 5);
    int lane = threadIdx.x & 31;
    int tilesN = N >> 4;
    int total = (M >> 4) * tilesN;
    if (wave >= total) return;              // wave-uniform, EXEC stays full
    int tm = wave / tilesN;
    int tn = wave - tm * tilesN;
    int half = lane >> 4;                   // 0: lanes 0-15, 1: lanes 16-31
    int l16  = lane & 15;
    int rowA = tm * 16 + l16;               // A: M = lane%16
    int colB = tn * 16 + l16;               // B: N = lane%16
    const _Float16* arow = A + (size_t)rowA * K;
    const _Float16* brow = B + (size_t)colB * K;

    v8f acc = {};
    for (int k0 = 0; k0 < K; k0 += 32) {
        // A 16x32: lane holds K = k0+half*8 + [0..7]  and  k0+16+half*8 + [0..7]
        v8h al = *(const v8h*)(arow + k0 + half * 8);
        v8h ah = *(const v8h*)(arow + k0 + 16 + half * 8);
        // B 32x16: lane holds K = k0+half*8 + [0..7]  and  k0+16+half*8 + [0..7]
        v8h bl = *(const v8h*)(brow + k0 + half * 8);
        v8h bh = *(const v8h*)(brow + k0 + 16 + half * 8);
        v16h av, bv;
        #pragma unroll
        for (int i = 0; i < 8; ++i) {
            av[i] = al[i]; av[i + 8] = ah[i];
            bv[i] = bl[i]; bv[i + 8] = bh[i];
        }
        acc = __builtin_amdgcn_wmma_f32_16x16x32_f16(
                false, av, false, bv, (short)0, acc, false, false);
    }

    int col = tn * 16 + l16;
    float bval = bias ? bias[col] : 0.0f;
    #pragma unroll
    for (int i = 0; i < 8; ++i) {
        int row = tm * 16 + (half << 3) + i;   // C: M = vgpr + half*8
        C[(size_t)row * N + col] = acc[i] + bval;
    }
}

// ---------------------------------------------------------------------------
// Cantor coordinate per row: max-sim reduce over 500 centroids, 8 ternary
// digits, quantized exactly to integer in [0,255].
// ---------------------------------------------------------------------------
__global__ void cantor_kernel(const float* __restrict__ sims,
                              const float* __restrict__ gwp,
                              int* __restrict__ c_int) {
    __shared__ float red[128];
    int n = blockIdx.x, tid = threadIdx.x;   // 128 threads
    float m = -1e30f;
    for (int c = tid; c < NCENT; c += 128)
        m = fmaxf(m, sims[(size_t)n * CPAD + c]);
    red[tid] = m;
    __syncthreads();
    for (int st = 64; st > 0; st >>= 1) {
        if (tid < st) red[tid] = fmaxf(red[tid], red[tid + st]);
        __syncthreads();
    }
    if (tid == 0) {
        float gd  = 1.0f - red[0];
        float pos = (float)n / (float)(SEQ_N - 1);
        float gw  = 1.0f / (1.0f + expf(-gwp[0]));
        float xc  = pos * (1.0f - gw) + gd * gw;
        xc = fminf(fmaxf(xc, 1e-6f), 1.0f - 1e-6f);
        float cant = 0.0f, factor = 0.5f;
        for (int i = 0; i < 8; ++i) {
            float xs = xc * 3.0f;
            float dg = floorf(xs);
            if (dg == 2.0f) cant += factor;
            xc = xs - dg;
            factor *= 0.5f;
        }
        cant = fminf(fmaxf(cant, 0.0f), 1.0f);
        int q = (int)(cant * 256.0f + 0.5f);
        c_int[n] = q > 255 ? 255 : q;
    }
}

// ---------------------------------------------------------------------------
// Exact 128-smallest selection per row by (dist,index) lexicographic order
// (matches jax.lax.top_k tie-breaking). Distances are discrete in [0,255].
// One 256-thread block per row.
// ---------------------------------------------------------------------------
__global__ void topk_kernel(const int* __restrict__ c_int,
                            int* __restrict__ routes) {
    __shared__ int hist[256];
    __shared__ int eqcnt[256];
    __shared__ int eqoff[256];
    __shared__ int s_t, s_less, s_need, s_slot;
    int n = blockIdx.x, tid = threadIdx.x;   // 256 threads
    int cn = c_int[n];
    hist[tid] = 0;
    __syncthreads();
    for (int j = tid; j < SEQ_N; j += 256) {
        int d = cn - c_int[j]; d = d < 0 ? -d : d;
        atomicAdd(&hist[d], 1);
    }
    __syncthreads();
    if (tid == 0) {
        int acc = 0, t = 0;
        for (t = 0; t < 256; ++t) {
            if (acc + hist[t] >= KWIN) break;
            acc += hist[t];
        }
        s_t = t; s_less = acc; s_need = KWIN - acc; s_slot = 0;
    }
    __syncthreads();
    int t = s_t, less = s_less, need = s_need;
    // rank dist==t entries by index: chunked counts + prefix
    int base = tid * 8, ec = 0;
    for (int j = base; j < base + 8; ++j) {
        int d = cn - c_int[j]; d = d < 0 ? -d : d;
        if (d == t) ec++;
    }
    eqcnt[tid] = ec;
    __syncthreads();
    if (tid == 0) {
        int a = 0;
        for (int i = 0; i < 256; ++i) { eqoff[i] = a; a += eqcnt[i]; }
    }
    __syncthreads();
    int eo = eqoff[tid], lr = 0;
    for (int j = base; j < base + 8; ++j) {
        int d = cn - c_int[j]; d = d < 0 ? -d : d;
        if (d < t) {
            int p = atomicAdd(&s_slot, 1);
            routes[(size_t)n * KWIN + p] = j;
        } else if (d == t) {
            int r = eo + lr; lr++;
            if (r < need) routes[(size_t)n * KWIN + less + r] = j;
        }
    }
}

// ---------------------------------------------------------------------------
// Gathered attention: one block per query n, 8 waves = 8 heads.
// qkv layout: [n][0:512)=Q, [512:1024)=K, [1024:1536)=V, head-major inside.
// ---------------------------------------------------------------------------
__global__ void attn_kernel(const float* __restrict__ qkv,
                            const int* __restrict__ routes,
                            _Float16* __restrict__ outh) {
    __shared__ int   r_lds[KWIN];
    __shared__ __align__(16) float q_lds[HEADS][HD];
    __shared__ float a_lds[HEADS][KWIN];
    int n    = blockIdx.x;
    int tid  = threadIdx.x;       // 256
    int h    = tid >> 5;          // wave id == head
    int lane = tid & 31;

    if (tid < KWIN) r_lds[tid] = routes[(size_t)n * KWIN + tid];
    int d0 = lane * 2;
    const float* qrow = qkv + (size_t)n * QKV3 + h * HD;
    q_lds[h][d0]     = qrow[d0];
    q_lds[h][d0 + 1] = qrow[d0 + 1];
    __syncthreads();

    const float scale = 0.125f;   // 1/sqrt(64)
    const float4* q4 = (const float4*)(&q_lds[h][0]);
    float sc[4];
    #pragma unroll
    for (int i = 0; i < 4; ++i) {
        int j = lane + 32 * i;
        int r = r_lds[j];
        const float4* k4 = (const float4*)(qkv + (size_t)r * QKV3 + DIM + h * HD);
        float acc = 0.0f;
        #pragma unroll
        for (int tq = 0; tq < 16; ++tq) {
            float4 qv = q4[tq], kv = k4[tq];
            acc += qv.x * kv.x + qv.y * kv.y + qv.z * kv.z + qv.w * kv.w;
        }
        sc[i] = acc * scale;
    }
    // softmax over 128 values (4 per lane)
    float m = fmaxf(fmaxf(sc[0], sc[1]), fmaxf(sc[2], sc[3]));
    for (int o = 16; o > 0; o >>= 1) m = fmaxf(m, __shfl_xor(m, o, 32));
    float p[4], s = 0.0f;
    #pragma unroll
    for (int i = 0; i < 4; ++i) { p[i] = expf(sc[i] - m); s += p[i]; }
    for (int o = 16; o > 0; o >>= 1) s += __shfl_xor(s, o, 32);
    float inv = 1.0f / s;
    #pragma unroll
    for (int i = 0; i < 4; ++i) a_lds[h][lane + 32 * i] = p[i] * inv;
    __syncthreads();

    // AV: each lane owns two of the 64 head dims
    float o0 = 0.0f, o1 = 0.0f;
    for (int j = 0; j < KWIN; ++j) {
        float a = a_lds[h][j];
        const float* vr = qkv + (size_t)r_lds[j] * QKV3 + 2 * DIM + h * HD;
        o0 += a * vr[d0];
        o1 += a * vr[d0 + 1];
    }
    outh[(size_t)n * DIM + h * HD + d0]     = (_Float16)o0;
    outh[(size_t)n * DIM + h * HD + d0 + 1] = (_Float16)o1;
}

// ---------------------------------------------------------------------------
extern "C" void kernel_launch(void* const* d_in, const int* in_sizes, int n_in,
                              void* d_out, int out_size, void* d_ws, size_t ws_size,
                              hipStream_t stream) {
    const float* x      = (const float*)d_in[0];
    const float* penta  = (const float*)d_in[1];
    const float* w_qkv  = (const float*)d_in[2];
    const float* b_qkv  = (const float*)d_in[3];
    const float* w_out  = (const float*)d_in[4];
    const float* b_out  = (const float*)d_in[5];
    const float* gw     = (const float*)d_in[6];
    float* out = (float*)d_out;

    // carve workspace (256B aligned slices)
    unsigned char* p = (unsigned char*)d_ws;
    auto carve = [&](size_t bytes) -> void* {
        void* r = (void*)p;
        p += (bytes + 255) & ~(size_t)255;
        return r;
    };
    _Float16* x_h    = (_Float16*)carve((size_t)SEQ_N * DIM * 2);
    _Float16* xn_h   = (_Float16*)carve((size_t)SEQ_N * DIM * 2);
    _Float16* wqkv_h = (_Float16*)carve((size_t)QKV3 * DIM * 2);
    _Float16* wout_h = (_Float16*)carve((size_t)DIM * DIM * 2);
    _Float16* cent_h = (_Float16*)carve((size_t)CPAD * DIM * 2);
    float*    qkv    = (float*)carve((size_t)SEQ_N * QKV3 * 4);
    float*    sims   = (float*)carve((size_t)SEQ_N * CPAD * 4);
    int*      c_int  = (int*)carve((size_t)SEQ_N * 4);
    int*      routes = (int*)carve((size_t)SEQ_N * KWIN * 4);
    _Float16* ao_h   = (_Float16*)carve((size_t)SEQ_N * DIM * 2);

    // 1) precision converts / prep
    cvt_f16_kernel<<<(SEQ_N * DIM + 255) / 256, 256, 0, stream>>>(x, x_h, SEQ_N * DIM);
    cvt_f16_kernel<<<(QKV3 * DIM + 255) / 256, 256, 0, stream>>>(w_qkv, wqkv_h, QKV3 * DIM);
    cvt_f16_kernel<<<(DIM * DIM + 255) / 256, 256, 0, stream>>>(w_out, wout_h, DIM * DIM);
    centroid_kernel<<<(CPAD * DIM + 255) / 256, 256, 0, stream>>>(penta, cent_h);
    rownorm_kernel<<<SEQ_N, 128, 0, stream>>>(x, xn_h);

    // 2) QKV GEMM: [2048,1536] = x_h[2048,512] * w_qkv[1536,512]^T + b
    {
        int tiles = (SEQ_N / 16) * (QKV3 / 16);        // 12288
        gemm_f16_wmma<<<tiles / 8, 256, 0, stream>>>(x_h, wqkv_h, b_qkv, qkv,
                                                     SEQ_N, QKV3, DIM);
    }
    // 3) sims GEMM: [2048,512] = xn_h * cent_h^T (cols >=500 are zero rows)
    {
        int tiles = (SEQ_N / 16) * (CPAD / 16);        // 4096
        gemm_f16_wmma<<<tiles / 8, 256, 0, stream>>>(xn_h, cent_h, nullptr, sims,
                                                     SEQ_N, CPAD, DIM);
    }
    // 4) cantor coordinate + exact top-k routing
    cantor_kernel<<<SEQ_N, 128, 0, stream>>>(sims, gw, c_int);
    topk_kernel<<<SEQ_N, 256, 0, stream>>>(c_int, routes);

    // 5) gathered attention
    attn_kernel<<<SEQ_N, 256, 0, stream>>>(qkv, routes, ao_h);

    // 6) out projection: [2048,512] = ao_h * w_out[512,512]^T + b_out
    {
        int tiles = (SEQ_N / 16) * (DIM / 16);         // 4096
        gemm_f16_wmma<<<tiles / 8, 256, 0, stream>>>(ao_h, wout_h, b_out, out,
                                                     SEQ_N, DIM, DIM);
    }
    (void)in_sizes; (void)n_in; (void)out_size; (void)ws_size;
}
